// MoE_8658654068958
// MI455X (gfx1250) — matrix operand order repaired and checked by reading, not compiled, not run on hardware
//
#include <hip/hip_runtime.h>

#define H    1024
#define IDIM 2816
#define NE   8
#define TTOK 4096

#define MT     32           // token tile per block (2 WMMA M-tiles)
#define IC     128          // I-chunk
#define KB     128          // K-block staged per iteration (GEMM1)
#define NICH   (IDIM / IC)  // 22
#define XS_LD  (H + 8)      // 1032 bf16 row stride
#define W_LD   (KB + 8)     // 136 bf16
#define ACT_LD (IC + 8)     // 136 bf16

typedef __bf16 bf16;
typedef bf16  v16bf __attribute__((ext_vector_type(16)));
typedef bf16  v8bf  __attribute__((ext_vector_type(8)));
typedef bf16  v4bf  __attribute__((ext_vector_type(4)));
typedef float v8f   __attribute__((ext_vector_type(8)));

#define SMEM_BYTES (MT * XS_LD * 2 + 2 * KB * W_LD * 2 + MT * ACT_LD * 2 + MT * 4 + MT * 4)

__device__ __forceinline__ v16bf ld16cat(const bf16* p0, const bf16* p1) {
  v8bf a = *(const v8bf*)p0;
  v8bf b = *(const v8bf*)p1;
  return __builtin_shufflevector(a, b, 0, 1, 2, 3, 4, 5, 6, 7, 8, 9, 10, 11, 12, 13, 14, 15);
}

__device__ __forceinline__ void st4bf(bf16* p, float a, float b, float c, float d) {
  v4bf v;
  v[0] = (bf16)a; v[1] = (bf16)b; v[2] = (bf16)c; v[3] = (bf16)d;
  *(v4bf*)p = v;   // single b64 LDS store
}

// ---------------- Router: logits, top-2 softmax, per-expert token lists ----------------
__global__ void __launch_bounds__(256) moe_gate(
    const float* __restrict__ x, const float* __restrict__ gw,
    float* __restrict__ gate_out, int* __restrict__ counts,
    int* __restrict__ tokList, float* __restrict__ pList) {
  const int lane = threadIdx.x & 31;
  const int wave = threadIdx.x >> 5;
  const int t = blockIdx.x * 8 + wave;

  float acc[NE];
#pragma unroll
  for (int e = 0; e < NE; ++e) acc[e] = 0.f;
  const float* xr = x + (size_t)t * H;
  for (int h = lane; h < H; h += 32) {
    float xv = xr[h];
    const float* g = gw + (size_t)h * NE;
#pragma unroll
    for (int e = 0; e < NE; ++e) acc[e] += xv * g[e];
  }
#pragma unroll
  for (int off = 16; off > 0; off >>= 1) {
#pragma unroll
    for (int e = 0; e < NE; ++e) acc[e] += __shfl_xor(acc[e], off, 32);
  }
  if (lane == 0) {
#pragma unroll
    for (int e = 0; e < NE; ++e) gate_out[(size_t)t * NE + e] = acc[e];
    int i1 = 0; float v1 = acc[0];
#pragma unroll
    for (int e = 1; e < NE; ++e) if (acc[e] > v1) { v1 = acc[e]; i1 = e; }
    int i2 = -1; float v2 = -3.4e38f;
#pragma unroll
    for (int e = 0; e < NE; ++e) if (e != i1 && acc[e] > v2) { v2 = acc[e]; i2 = e; }
    float d  = expf(v2 - v1);        // stable 2-way softmax over the top-2
    float p1 = 1.f / (1.f + d);
    float p2 = d / (1.f + d);
    int s1 = atomicAdd(&counts[i1], 1);
    tokList[i1 * TTOK + s1] = t; pList[i1 * TTOK + s1] = p1;
    int s2 = atomicAdd(&counts[i2], 1);
    tokList[i2 * TTOK + s2] = t; pList[i2 * TTOK + s2] = p2;
  }
}

// ---------------- Expert kernel: 32-token tile per block, one expert ----------------
__global__ void __launch_bounds__(256) moe_expert(
    const float* __restrict__ x, const float* __restrict__ Wg,
    const float* __restrict__ Wu, const float* __restrict__ Wd,
    const int* __restrict__ counts, const int* __restrict__ tokList,
    const float* __restrict__ pList, float* __restrict__ out) {
  const int e = blockIdx.y;
  const int cnt = counts[e];
  const int tile = blockIdx.x;
  if (tile * MT >= cnt) return;

  extern __shared__ char smem[];
  bf16* Xs  = (bf16*)smem;                                   // [MT][XS_LD]
  char* wrg = smem + MT * XS_LD * 2;
  bf16* WgS = (bf16*)wrg;                                    // [IC][W_LD] (N-major, K contiguous)
  bf16* WuS = (bf16*)(wrg + KB * W_LD * 2);
  bf16* WdS = (bf16*)wrg;                                    // phase-2 alias
  char* p2  = wrg + 2 * KB * W_LD * 2;
  bf16*  actS  = (bf16*)p2;                                  // [MT][ACT_LD]
  int*   tokS  = (int*)(p2 + MT * ACT_LD * 2);
  float* probS = (float*)(tokS + MT);

  const int tid  = threadIdx.x;
  const int lane = tid & 31;
  const int w    = tid >> 5;        // wave 0..7
  const int nL   = lane & 15;       // tile column / A row
  const int half = lane >> 4;
  const int kloA = half * 8;        // A: two 8-elem runs at +0 / +16
  const int kbB  = half * 8;        // B: two 8-elem runs at +0 / +16

  if (tid < MT) {
    int slot = tile * MT + tid;
    int tk = (slot < cnt) ? tokList[e * TTOK + slot] : -1;
    tokS[tid]  = tk;
    probS[tid] = (slot < cnt) ? pList[e * TTOK + slot] : 0.f;
  }
  __syncthreads();

  // gather + convert X rows -> bf16 LDS (float4 loads, packed b64 stores)
  for (int idx = tid; idx < MT * (H / 4); idx += 256) {
    int m = idx >> 8, h4 = idx & (H / 4 - 1);
    int tk = tokS[m];
    bf16* dst = Xs + m * XS_LD + h4 * 4;
    if (tk >= 0) {
      float4 v = ((const float4*)(x + (size_t)tk * H))[h4];
      st4bf(dst, v.x, v.y, v.z, v.w);
    } else {
      st4bf(dst, 0.f, 0.f, 0.f, 0.f);
    }
  }

  const v8f vz = {0.f, 0.f, 0.f, 0.f, 0.f, 0.f, 0.f, 0.f};
  v8f acc[16];                       // [nb][mt] tiles, 32x1024 f32 block tile
#pragma unroll
  for (int i = 0; i < 16; ++i) acc[i] = vz;

  const size_t wgeBase = (size_t)e * H * IDIM;
  const size_t wdeBase = (size_t)e * IDIM * H;

  for (int ic = 0; ic < NICH; ++ic) {
    const int i0 = ic * IC;

    // ---- GEMM1: g,u[MT,IC] = X[MT,H] @ {Wg,Wu}[H, i0:i0+IC] ----
    v8f gC0 = vz, gC1 = vz, uC0 = vz, uC1 = vz;
    for (int kb = 0; kb < H / KB; ++kb) {
      __syncthreads();  // staged-LDS consumers from previous iteration are done
      // stage [KB x IC] transposed -> [IC][KB], 4 k-values per thread, packed store
      for (int idx = tid; idx < (KB / 4) * IC; idx += 256) {
        int c = idx & (IC - 1);
        int k = (idx >> 7) << 2;
        size_t ga = wgeBase + (size_t)(kb * KB + k) * IDIM + (i0 + c);
        st4bf(WgS + c * W_LD + k, Wg[ga], Wg[ga + IDIM], Wg[ga + 2 * IDIM], Wg[ga + 3 * IDIM]);
        st4bf(WuS + c * W_LD + k, Wu[ga], Wu[ga + IDIM], Wu[ga + 2 * IDIM], Wu[ga + 3 * IDIM]);
        if (kb + 1 < H / KB) {
          __builtin_prefetch(Wg + ga + (size_t)KB * IDIM, 0, 0);
          __builtin_prefetch(Wu + ga + (size_t)KB * IDIM, 0, 0);
        }
      }
      __syncthreads();
      const bf16* xrow0 = Xs + nL * XS_LD + kb * KB + kloA;         // M-tile 0
      const bf16* xrow1 = xrow0 + 16 * XS_LD;                       // M-tile 1
      const bf16* grow  = WgS + (16 * w + nL) * W_LD + kbB;
      const bf16* urow  = WuS + (16 * w + nL) * W_LD + kbB;
#pragma unroll
      for (int k0 = 0; k0 < KB; k0 += 32) {
        v16bf A0 = ld16cat(xrow0 + k0, xrow0 + k0 + 16);
        v16bf A1 = ld16cat(xrow1 + k0, xrow1 + k0 + 16);
        v16bf Bg = ld16cat(grow + k0, grow + k0 + 16);
        v16bf Bu = ld16cat(urow + k0, urow + k0 + 16);
        gC0 = __builtin_amdgcn_wmma_f32_16x16x32_bf16(false, A0, false, Bg, (short)0, gC0, false, false);
        gC1 = __builtin_amdgcn_wmma_f32_16x16x32_bf16(false, A1, false, Bg, (short)0, gC1, false, false);
        uC0 = __builtin_amdgcn_wmma_f32_16x16x32_bf16(false, A0, false, Bu, (short)0, uC0, false, false);
        uC1 = __builtin_amdgcn_wmma_f32_16x16x32_bf16(false, A1, false, Bu, (short)0, uC1, false, false);
      }
    }

    // SwiGLU epilogue (f32) -> actS (bf16). C/D layout: row = v + 8*half (+16*mt), col = nL.
#pragma unroll
    for (int v = 0; v < 8; ++v) {
      int m = v + (half ? 8 : 0);
      float g0 = gC0[v], u0 = uC0[v];
      float s0 = (g0 / (1.f + expf(-g0))) * u0;
      actS[m * ACT_LD + 16 * w + nL] = (bf16)s0;
      float g1 = gC1[v], u1 = uC1[v];
      float s1 = (g1 / (1.f + expf(-g1))) * u1;
      actS[(m + 16) * ACT_LD + 16 * w + nL] = (bf16)s1;
    }

    // ---- GEMM2: acc[MT,1024] += act[MT,IC] @ Wd[i0:i0+IC, :] ----
    for (int nb = 0; nb < 8; ++nb) {
      __syncthreads();  // actS complete / WdS consumers done
      for (int idx = tid; idx < (IC / 4) * 128; idx += 256) {
        int c = idx & 127;
        int k = (idx >> 7) << 2;
        size_t ga = wdeBase + (size_t)(i0 + k) * H + nb * 128 + c;
        st4bf(WdS + c * W_LD + k, Wd[ga], Wd[ga + H], Wd[ga + 2 * H], Wd[ga + 3 * H]);
      }
      __syncthreads();
      const bf16* arow0 = actS + nL * ACT_LD + kloA;
      const bf16* arow1 = arow0 + 16 * ACT_LD;
      const bf16* brow  = WdS + (16 * w + nL) * W_LD + kbB;
      v8f c0 = acc[nb * 2 + 0];
      v8f c1 = acc[nb * 2 + 1];
#pragma unroll
      for (int k0 = 0; k0 < IC; k0 += 32) {
        v16bf A0 = ld16cat(arow0 + k0, arow0 + k0 + 16);
        v16bf A1 = ld16cat(arow1 + k0, arow1 + k0 + 16);
        v16bf B  = ld16cat(brow + k0, brow + k0 + 16);
        c0 = __builtin_amdgcn_wmma_f32_16x16x32_bf16(false, A0, false, B, (short)0, c0, false, false);
        c1 = __builtin_amdgcn_wmma_f32_16x16x32_bf16(false, A1, false, B, (short)0, c1, false, false);
      }
      acc[nb * 2 + 0] = c0;
      acc[nb * 2 + 1] = c1;
    }
  }

  // scale by routing prob, accumulate into output (2 commutative f32 atomics per element)
#pragma unroll
  for (int nb = 0; nb < 8; ++nb) {
    int hcol = nb * 128 + 16 * w + nL;
#pragma unroll
    for (int mt = 0; mt < 2; ++mt) {
#pragma unroll
      for (int v = 0; v < 8; ++v) {
        int m = v + (half ? 8 : 0) + 16 * mt;
        int tk = tokS[m];
        if (tk >= 0)
          unsafeAtomicAdd(&out[(size_t)tk * H + hcol], acc[nb * 2 + mt][v] * probS[m]);
      }
    }
  }
}

extern "C" void kernel_launch(void* const* d_in, const int* in_sizes, int n_in,
                              void* d_out, int out_size, void* d_ws, size_t ws_size,
                              hipStream_t stream) {
  const float* x  = (const float*)d_in[0];
  const float* gw = (const float*)d_in[1];
  const float* Wg = (const float*)d_in[2];
  const float* Wu = (const float*)d_in[3];
  const float* Wd = (const float*)d_in[4];

  float* out      = (float*)d_out;                       // [T, H]
  float* gate_out = (float*)d_out + (size_t)TTOK * H;    // [T, E]

  int*   counts  = (int*)d_ws;
  int*   tokList = (int*)((char*)d_ws + 64);                           // [E][T]
  float* pList   = (float*)((char*)d_ws + 64 + (size_t)NE * TTOK * 4); // [E][T]

  (void)in_sizes; (void)n_in; (void)out_size; (void)ws_size;

  hipMemsetAsync(d_out, 0, (size_t)TTOK * H * sizeof(float), stream);
  hipMemsetAsync(d_ws, 0, 64, stream);

  moe_gate<<<TTOK / 8, 256, 0, stream>>>(x, gw, gate_out, counts, tokList, pList);

  hipFuncSetAttribute((const void*)moe_expert,
                      hipFuncAttributeMaxDynamicSharedMemorySize, SMEM_BYTES);

  dim3 grid(TTOK / MT, NE);
  moe_expert<<<grid, 256, SMEM_BYTES, stream>>>(x, Wg, Wu, Wd, counts, tokList, pList, out);
}
